// HybridTransformerBlock_49392123904225
// MI455X (gfx1250) — matrix operand order repaired and checked by reading, not compiled, not run on hardware
//
#include <hip/hip_runtime.h>
#include <hip/hip_bf16.h>
#include <math.h>

// ---------------------------------------------------------------------------
// HybridTransformerBlock on MI455X (gfx1250): bf16 WMMA + TDM-fed LDS GEMMs.
//  B=4 S=2048 D=1024 H=16 DH=32 DP=64 DFF=4096, M = B*S = 8192.
// ---------------------------------------------------------------------------

typedef __hip_bfloat16 bf16_t;
typedef __attribute__((ext_vector_type(16))) __bf16 v16bf;
typedef __attribute__((ext_vector_type(8)))  float  v8f;
typedef __attribute__((ext_vector_type(4)))  unsigned int v4u;
typedef __attribute__((ext_vector_type(8)))  int    v8i_t;
typedef __attribute__((ext_vector_type(4)))  int    v4i_t;

union Frag32B { v16bf v; uint4 q[2]; };

// ---- fragment loaders (global or LDS source; ISA 16-bit layouts) ----------
// A 16x32: lanes 0-15 -> M=lane, K={0..7,16..23}; lanes 16-31 -> K={8..15,24..31}
__device__ inline v16bf load_afrag(const bf16_t* tile, int ld) {
  const int lane = threadIdx.x & 31;
  const bf16_t* p = tile + (size_t)(lane & 15) * ld + ((lane >> 4) << 3);
  Frag32B f;
  f.q[0] = *(const uint4*)(p);
  f.q[1] = *(const uint4*)(p + 16);
  return f.v;
}
// B 32x16 from transposed weights Bt[N][K]: lane = column, K={0..7,16..23} / {8..15,24..31}
__device__ inline v16bf load_bfrag(const bf16_t* tile, int ld) {
  const int lane = threadIdx.x & 31;
  const bf16_t* p = tile + (size_t)(lane & 15) * ld + ((lane >> 4) << 3);
  Frag32B f;
  f.q[0] = *(const uint4*)(p);
  f.q[1] = *(const uint4*)(p + 16);
  return f.v;
}
// LDS variants with fixed ld = 64 (the staged K-chunk width)
__device__ inline v16bf lds_afrag(const bf16_t* tile) {
  const int lane = threadIdx.x & 31;
  const bf16_t* p = tile + (lane & 15) * 64 + ((lane >> 4) << 3);
  Frag32B f;
  f.q[0] = *(const uint4*)(p);
  f.q[1] = *(const uint4*)(p + 16);
  return f.v;
}
__device__ inline v16bf lds_bfrag(const bf16_t* tile) {
  const int lane = threadIdx.x & 31;
  const bf16_t* p = tile + (lane & 15) * 64 + ((lane >> 4) << 3);
  Frag32B f;
  f.q[0] = *(const uint4*)(p);
  f.q[1] = *(const uint4*)(p + 16);
  return f.v;
}

__device__ inline v8f wmma_bf16f32(v16bf a, v16bf b, v8f c) {
  return __builtin_amdgcn_wmma_f32_16x16x32_bf16(false, a, false, b,
                                                 (short)0, c, false, false);
}

__device__ inline float gelu_exact(float x) {
  return 0.5f * x * (1.0f + erff(x * 0.7071067811865476f));
}

// ---------------------------------------------------------------------------
// Tensor Data Mover: 2D tile (tile_rows x tile_k bf16, row ld = `ld` elems)
// from global -> LDS (packed row-major, ld = tile_k). D# per ISA ch.8.
// Wave-level op (EXEC ignored): call from exactly one wave per block.
// 6-arg builtin form: (g0 u32x4, g1 i32x8, g2 i32x4, g3 i32x4, i32x8, cpol)
// ---------------------------------------------------------------------------
__device__ inline void tdm_load_2d(unsigned lds_byte_off, const void* gptr,
                                   unsigned tile_k, unsigned tile_rows,
                                   unsigned ld) {
  unsigned long long ga = (unsigned long long)(size_t)gptr;
  v4u g0;
  g0[0] = 1u;                                  // count=1 (valid), user mode
  g0[1] = lds_byte_off;                        // D#.lds_addr
  g0[2] = (unsigned)ga;                        // global_addr[31:0] (bytes)
  g0[3] = (unsigned)((ga >> 32) & 0x01ffffffu) // global_addr[56:32]
          | (2u << 30);                        // type = 2 ("image")
  v8i_t g1;
  g1[0] = (int)(1u << 16);                     // wg_mask=0 | data_size=1 (2B)
  g1[1] = (int)(tile_k << 16);                 // tensor_dim0[15:0]
  g1[2] = (int)((tile_rows & 0xffffu) << 16);  // dim0 hi=0 | tensor_dim1[15:0]
  g1[3] = (int)(tile_k << 16);                 // dim1 hi=0 | tile_dim0
  g1[4] = (int)(tile_rows & 0xffffu);          // tile_dim1 | tile_dim2=0
  g1[5] = (int)ld;                             // tensor_dim0_stride[31:0] (elems)
  g1[6] = 0;                                   // stride hi | dim1_stride lo
  g1[7] = 0;
  v4i_t z4 = {0, 0, 0, 0};
  v8i_t z8 = {0, 0, 0, 0, 0, 0, 0, 0};
  __builtin_amdgcn_tensor_load_to_lds(g0, g1, z4, z4, z8, 0);
}

// ---------------------------------------------------------------------------
// One-time weight prep: f32 [K][N] -> bf16 transposed [N][K]
// ---------------------------------------------------------------------------
__global__ void transpose_bf16_kernel(const float* __restrict__ src,
                                      bf16_t* __restrict__ dst, int K, int N) {
  size_t id = (size_t)blockIdx.x * blockDim.x + threadIdx.x;
  if (id >= (size_t)K * N) return;
  int n = (int)(id / K);
  int k = (int)(id % K);
  dst[id] = __float2bfloat16(src[(size_t)k * N + n]);
}

// A[H][DH_i][DP_p][DH_j] (f32) -> At[h][(i*32+j)][p] (bf16)
__global__ void convert_A_kernel(const float* __restrict__ A,
                                 bf16_t* __restrict__ At) {
  int id = blockIdx.x * blockDim.x + threadIdx.x;  // < 16*65536
  int j = id & 31;
  int p = (id >> 5) & 63;
  int i = (id >> 11) & 31;
  int h = id >> 16;
  At[(size_t)h * 65536 + (size_t)(i * 32 + j) * 64 + p] = __float2bfloat16(A[id]);
}

// ---------------------------------------------------------------------------
// LayerNorm over D=1024, one block per row, bf16 output
// ---------------------------------------------------------------------------
__global__ __launch_bounds__(256) void ln_kernel(const float* __restrict__ x,
                                                 const float* __restrict__ g,
                                                 const float* __restrict__ beta,
                                                 bf16_t* __restrict__ out) {
  const int D = 1024;
  const size_t row = blockIdx.x;
  const float* xr = x + row * D;
  float v[4], s = 0.f, s2 = 0.f;
#pragma unroll
  for (int i = 0; i < 4; ++i) {
    v[i] = xr[threadIdx.x + i * 256];
    s += v[i];
    s2 += v[i] * v[i];
  }
#pragma unroll
  for (int o = 16; o > 0; o >>= 1) {
    s  += __shfl_down(s,  o, 32);
    s2 += __shfl_down(s2, o, 32);
  }
  __shared__ float rs[8], rs2[8];
  __shared__ float mean_s, rstd_s;
  const int w = threadIdx.x >> 5, lane = threadIdx.x & 31;
  if (lane == 0) { rs[w] = s; rs2[w] = s2; }
  __syncthreads();
  if (threadIdx.x == 0) {
    float t = 0.f, t2 = 0.f;
#pragma unroll
    for (int i = 0; i < 8; ++i) { t += rs[i]; t2 += rs2[i]; }
    float mean = t * (1.0f / D);
    float var  = t2 * (1.0f / D) - mean * mean;
    mean_s = mean;
    rstd_s = rsqrtf(var + 1e-5f);
  }
  __syncthreads();
  const float mean = mean_s, rstd = rstd_s;
#pragma unroll
  for (int i = 0; i < 4; ++i) {
    int c = threadIdx.x + i * 256;
    out[row * D + c] = __float2bfloat16((v[i] - mean) * rstd * g[c] + beta[c]);
  }
}

// ---------------------------------------------------------------------------
// TDM-fed bf16 WMMA GEMM: C = act(A[M,K] @ Bt[N,K]^T + bias) (+ resid)
// block = 256 thr = 8 waves; wave tile 32x32; block tile 64(M) x 128(N).
// Double-buffered 64-wide K chunks staged to LDS by the Tensor Data Mover,
// synchronized with s_wait_tensorcnt; math waves feed WMMA from ds_load_b128.
// ---------------------------------------------------------------------------
__global__ __launch_bounds__(256) void gemm_bf16_kernel(
    const bf16_t* __restrict__ A, const bf16_t* __restrict__ Bt,
    const float* __restrict__ bias, const float* __restrict__ resid,
    float* __restrict__ Cf, bf16_t* __restrict__ Cb,
    int M, int N, int K, int act) {
  __shared__ bf16_t As[2][64][64];    // 16 KB
  __shared__ bf16_t Bs[2][128][64];   // 32 KB
  const int lane = threadIdx.x & 31;
  const int w = threadIdx.x >> 5;
  const int wm = w >> 2, wn = w & 3;
  const int m_base = blockIdx.y * 64;
  const int n_base = blockIdx.x * 128;
  const bool issuer = (w == 0);       // wave-uniform: one DMA pair per chunk

  if (issuer) {
    tdm_load_2d((unsigned)(size_t)&As[0][0][0], A + (size_t)m_base * K, 64, 64, K);
    tdm_load_2d((unsigned)(size_t)&Bs[0][0][0], Bt + (size_t)n_base * K, 64, 128, K);
  }

  v8f c[2][2] = {};
  int buf = 0;
  for (int kc = 0; kc < K; kc += 64) {
    if (issuer) {
      if (kc + 64 < K) {  // prefetch next chunk into the other buffer
        tdm_load_2d((unsigned)(size_t)&As[buf ^ 1][0][0],
                    A + (size_t)m_base * K + (kc + 64), 64, 64, K);
        tdm_load_2d((unsigned)(size_t)&Bs[buf ^ 1][0][0],
                    Bt + (size_t)n_base * K + (kc + 64), 64, 128, K);
        // in-order TENSORcnt: <=2 outstanding ==> current buffer complete
        __builtin_amdgcn_s_wait_tensorcnt(2);
      } else {
        __builtin_amdgcn_s_wait_tensorcnt(0);
      }
    }
    __syncthreads();
#pragma unroll
    for (int kk = 0; kk < 64; kk += 32) {
      v16bf a0 = lds_afrag(&As[buf][wm * 32 +  0][kk]);
      v16bf a1 = lds_afrag(&As[buf][wm * 32 + 16][kk]);
      v16bf b0 = lds_bfrag(&Bs[buf][wn * 32 +  0][kk]);
      v16bf b1 = lds_bfrag(&Bs[buf][wn * 32 + 16][kk]);
      c[0][0] = wmma_bf16f32(a0, b0, c[0][0]);
      c[0][1] = wmma_bf16f32(a0, b1, c[0][1]);
      c[1][0] = wmma_bf16f32(a1, b0, c[1][0]);
      c[1][1] = wmma_bf16f32(a1, b1, c[1][1]);
    }
    __syncthreads();
    buf ^= 1;
  }

  // Epilogue: C/D layout = lane<16: col=lane, rows 0..7; lane>=16: rows 8..15
  const int col = lane & 15;
  const int r0 = (lane >> 4) * 8;
  const int wm_base = m_base + wm * 32;
  const int wn_base = n_base + wn * 32;
#pragma unroll
  for (int mi = 0; mi < 2; ++mi) {
#pragma unroll
    for (int ni = 0; ni < 2; ++ni) {
      const int n = wn_base + ni * 16 + col;
      const float bv = bias ? bias[n] : 0.0f;
#pragma unroll
      for (int r = 0; r < 8; ++r) {
        const int m = wm_base + mi * 16 + r0 + r;
        float vv = c[mi][ni][r] + bv;
        if (act == 1) vv = gelu_exact(vv);
        const size_t idx = (size_t)m * N + n;
        if (resid) vv += resid[idx];
        if (Cf) Cf[idx] = vv;
        if (Cb) Cb[idx] = __float2bfloat16(vv);
      }
    }
  }
}

// ---------------------------------------------------------------------------
// Fused MPS scan: one block per (b,h). 8 waves build G_t[32x32] for a
// 16-timestep chunk in LDS via WMMA (v[16,64] @ A_h[64,1024]); wave 0 then
// runs the strictly-sequential tanh recurrence h_t = tanh(h_{t-1} @ G_t).
// ---------------------------------------------------------------------------
__global__ __launch_bounds__(256) void scan_kernel(
    const bf16_t* __restrict__ v_bf,   // [B*S, 1024] = [b,s][h*64+p]
    const bf16_t* __restrict__ At,     // [16][1024][64]
    bf16_t* __restrict__ hs_out) {     // [B*S, 512] = [b,s][h*32+j]
  __shared__ float Gs[16 * 1024];      // 64 KB: [t in chunk][i*32+j]
  const int b = blockIdx.x >> 4;
  const int h = blockIdx.x & 15;
  const int w = threadIdx.x >> 5;
  const int lane = threadIdx.x & 31;
  const bf16_t* Ah = At + (size_t)h * 65536;
  float hj = 0.17677669529663687f;     // 1/sqrt(DH)

  for (int t0 = 0; t0 < 2048; t0 += 16) {
    const bf16_t* vbase = v_bf + (size_t)(b * 2048 + t0) * 1024 + h * 64;
    __builtin_prefetch(vbase + 16 * 1024, 0, 3);  // next chunk's v rows
    v16bf a0 = load_afrag(vbase + 0, 1024);   // K = 0..31  (of DP=64)
    v16bf a1 = load_afrag(vbase + 32, 1024);  // K = 32..63
#pragma unroll
    for (int nt = 0; nt < 8; ++nt) {
      const int n0 = (w * 8 + nt) * 16;
      v16bf b0 = load_bfrag(Ah + (size_t)n0 * 64 + 0, 64);
      v16bf b1 = load_bfrag(Ah + (size_t)n0 * 64 + 32, 64);
      v8f c = {};
      c = wmma_bf16f32(a0, b0, c);
      c = wmma_bf16f32(a1, b1, c);
      const int col = lane & 15;
      const int r0 = (lane >> 4) * 8;
#pragma unroll
      for (int r = 0; r < 8; ++r)
        Gs[(r0 + r) * 1024 + n0 + col] = c[r];
    }
    __syncthreads();
    if (w == 0) {
      for (int t = 0; t < 16; ++t) {
        float m = 0.0f;
#pragma unroll
        for (int i = 0; i < 32; ++i)
          m += __shfl(hj, i, 32) * Gs[t * 1024 + i * 32 + lane];
        hj = tanhf(m);
        hs_out[(size_t)(b * 2048 + t0 + t) * 512 + h * 32 + lane] =
            __float2bfloat16(hj);
      }
    }
    __syncthreads();
  }
}

// ---------------------------------------------------------------------------
// Host-side orchestration
// ---------------------------------------------------------------------------
extern "C" void kernel_launch(void* const* d_in, const int* in_sizes, int n_in,
                              void* d_out, int out_size, void* d_ws, size_t ws_size,
                              hipStream_t stream) {
  (void)in_sizes; (void)n_in; (void)out_size; (void)ws_size;
  const int M = 8192;          // B*S
  const int D = 1024, DFF = 4096;

  const float* x     = (const float*)d_in[0];
  const float* ln1_g = (const float*)d_in[1];
  const float* ln1_b = (const float*)d_in[2];
  const float* W_in  = (const float*)d_in[3];
  const float* b_in  = (const float*)d_in[4];
  const float* Amat  = (const float*)d_in[5];
  const float* W_out = (const float*)d_in[6];
  const float* b_out = (const float*)d_in[7];
  const float* ln2_g = (const float*)d_in[8];
  const float* ln2_b = (const float*)d_in[9];
  const float* W1    = (const float*)d_in[10];
  const float* b1    = (const float*)d_in[11];
  const float* W2    = (const float*)d_in[12];
  const float* b2    = (const float*)d_in[13];

  char* ws = (char*)d_ws;
  size_t off = 0;
  auto alloc = [&](size_t bytes) -> void* {
    void* p = ws + off;
    off += (bytes + 255) & ~(size_t)255;
    return p;
  };
  bf16_t* h_bf   = (bf16_t*)alloc((size_t)M * D * 2);
  bf16_t* v_bf   = (bf16_t*)alloc((size_t)M * D * 2);
  bf16_t* hs_bf  = (bf16_t*)alloc((size_t)M * 512 * 2);
  float*  x1     = (float*) alloc((size_t)M * D * 4);
  bf16_t* h2_bf  = (bf16_t*)alloc((size_t)M * D * 2);
  bf16_t* f_bf   = (bf16_t*)alloc((size_t)M * DFF * 2);
  bf16_t* Wt_in  = (bf16_t*)alloc((size_t)1024 * 1024 * 2);
  bf16_t* Wt_out = (bf16_t*)alloc((size_t)1024 * 512 * 2);
  bf16_t* Wt1    = (bf16_t*)alloc((size_t)4096 * 1024 * 2);
  bf16_t* Wt2    = (bf16_t*)alloc((size_t)1024 * 4096 * 2);
  bf16_t* At     = (bf16_t*)alloc((size_t)16 * 65536 * 2);

  // ---- one-time weight conversion / transpose ----
  transpose_bf16_kernel<<<(1024 * 1024 + 255) / 256, 256, 0, stream>>>(W_in, Wt_in, 1024, 1024);
  transpose_bf16_kernel<<<(512 * 1024 + 255) / 256, 256, 0, stream>>>(W_out, Wt_out, 512, 1024);
  transpose_bf16_kernel<<<(1024 * 4096 + 255) / 256, 256, 0, stream>>>(W1, Wt1, 1024, 4096);
  transpose_bf16_kernel<<<(4096 * 1024 + 255) / 256, 256, 0, stream>>>(W2, Wt2, 4096, 1024);
  convert_A_kernel<<<(16 * 65536) / 256, 256, 0, stream>>>(Amat, At);

  // ---- branch 1: LN1 -> W_in GEMM -> MPS scan -> W_out GEMM + residual ----
  ln_kernel<<<M, 256, 0, stream>>>(x, ln1_g, ln1_b, h_bf);
  gemm_bf16_kernel<<<dim3(1024 / 128, M / 64), 256, 0, stream>>>(
      h_bf, Wt_in, b_in, nullptr, nullptr, v_bf, M, 1024, 1024, 0);
  scan_kernel<<<64, 256, 0, stream>>>(v_bf, At, hs_bf);
  gemm_bf16_kernel<<<dim3(1024 / 128, M / 64), 256, 0, stream>>>(
      hs_bf, Wt_out, b_out, x, x1, nullptr, M, 1024, 512, 0);

  // ---- branch 2: LN2 -> FFN (GELU) + residual -> d_out ----
  ln_kernel<<<M, 256, 0, stream>>>(x1, ln2_g, ln2_b, h2_bf);
  gemm_bf16_kernel<<<dim3(4096 / 128, M / 64), 256, 0, stream>>>(
      h2_bf, Wt1, b1, nullptr, nullptr, f_bf, M, 4096, 1024, 1);
  gemm_bf16_kernel<<<dim3(1024 / 128, M / 64), 256, 0, stream>>>(
      f_bf, Wt2, b2, x1, (float*)d_out, nullptr, M, 1024, 4096, 0);
}
